// MoEPlusPlusLayer_24713241821318
// MI455X (gfx1250) — matrix operand order repaired and checked by reading, not compiled, not run on hardware
//
#include <hip/hip_runtime.h>
#include <hip/hip_bf16.h>

// ---------------- problem constants (match reference) ----------------
#define NTOK 8192            // B*S = 4*2048
#define HD   1024            // hidden
#define ID   4096            // intermediate
#define NE   8               // experts
#define NSLOT (2*NTOK)       // top-2 assignments per token
#define TILES_PER_E 256      // NTOK/32 worst-case M tiles per expert

typedef unsigned short u16;
typedef __attribute__((ext_vector_type(16))) __bf16 v16bf;
typedef __attribute__((ext_vector_type(8)))  float  v8f;

union Frag { v16bf v; uint4 q[2]; };

__device__ __forceinline__ u16 f2bf(float f) {
    unsigned int u = __float_as_uint(f);
    unsigned int r = (u + 0x7FFFu + ((u >> 16) & 1u)) >> 16;   // RNE
    return (u16)r;
}

// XOR-swizzled 16B-chunk offset (in u16 units) within a padded Bs row:
// chunk c (k = 8c..8c+7) lands at chunk c ^ ((n>>4)&3). Constant across the
// 16 lanes of a WMMA B-fragment read (n>>4 fixed there) -> loads stay
// conflict-free; stores get spread across banks.
__device__ __forceinline__ int bofs(int n, int c) {
    return ((c ^ ((n >> 4) & 3)) << 3);
}

// ---------------- workspace layout (bytes) ----------------
static const size_t OFF_XBF  = 0;
static const size_t OFF_W1BF = OFF_XBF  + (size_t)NTOK * HD * 2;
static const size_t OFF_W2BF = OFF_W1BF + (size_t)NE * HD * ID * 2;
static const size_t OFF_HBUF = OFF_W2BF + (size_t)NE * ID * HD * 2;
static const size_t OFF_YBUF = OFF_HBUF + (size_t)NSLOT * ID * 2;
static const size_t OFF_GATE = OFF_YBUF + (size_t)NSLOT * HD * 4;
static const size_t OFF_CNT  = OFF_GATE + (size_t)NSLOT * 4;
static const size_t OFF_LIST = OFF_CNT  + 256;

// ---------------- tiny init: reset expert counters each call ----------------
__global__ void init_cnt_kernel(int* __restrict__ cnt) {
    if (threadIdx.x < NE) cnt[threadIdx.x] = 0;
}

// ---------------- fp32 -> bf16 bulk convert ----------------
__global__ __launch_bounds__(256)
void convert_bf16_kernel(const float* __restrict__ src, u16* __restrict__ dst, size_t n4) {
    size_t i = (size_t)blockIdx.x * blockDim.x + threadIdx.x;
    size_t stride = (size_t)gridDim.x * blockDim.x;
    for (; i < n4; i += stride) {
        float4 v = ((const float4*)src)[i];
        union { u16 s[4]; uint2 u; } o;
        o.s[0] = f2bf(v.x); o.s[1] = f2bf(v.y);
        o.s[2] = f2bf(v.z); o.s[3] = f2bf(v.w);
        ((uint2*)dst)[i] = o.u;
    }
}

// ---------------- routing: logits, confidence MLP, softmax, top-k, compaction ----
__global__ __launch_bounds__(256)
void routing_kernel(const float* __restrict__ x,   const float* __restrict__ Wr,
                    const float* __restrict__ br,  const float* __restrict__ Wc1,
                    const float* __restrict__ bc1, const float* __restrict__ Wc2,
                    const float* __restrict__ bc2,
                    float* __restrict__ outSW, int* __restrict__ outSI,
                    float* __restrict__ outConf, float* __restrict__ outLG,
                    float* __restrict__ gates, int* __restrict__ cnt,
                    int* __restrict__ lists)
{
    __shared__ __align__(16) float xs[16][HD];   // 64 KB of the 320 KB WGP LDS
    __shared__ float lg[16][8];
    __shared__ float red[16][8];

    const int t0 = blockIdx.x * 16;
    const int tid = threadIdx.x;

    for (int i = tid; i < 16 * HD / 4; i += 256)
        ((float4*)&xs[0][0])[i] = ((const float4*)(x + (size_t)t0 * HD))[i];
    __syncthreads();

    // confidence layer 1: each thread owns cols c and c+256 for all 16 tokens
    const int c = tid;
    float a0[16], a1[16];
#pragma unroll
    for (int m = 0; m < 16; ++m) { a0[m] = bc1[c]; a1[m] = bc1[c + 256]; }
    for (int h = 0; h < HD; ++h) {
        const float w0 = Wc1[(size_t)h * 512 + c];
        const float w1 = Wc1[(size_t)h * 512 + c + 256];
#pragma unroll
        for (int m = 0; m < 16; ++m) {
            const float xv = xs[m][h];
            a0[m] = fmaf(w0, xv, a0[m]);
            a1[m] = fmaf(w1, xv, a1[m]);
        }
    }
    const float wc0 = Wc2[c], wc1v = Wc2[c + 256];
    float p[16];
#pragma unroll
    for (int m = 0; m < 16; ++m)
        p[m] = fmaxf(a0[m], 0.f) * wc0 + fmaxf(a1[m], 0.f) * wc1v;
#pragma unroll
    for (int off = 16; off > 0; off >>= 1)
#pragma unroll
        for (int m = 0; m < 16; ++m)
            p[m] += __shfl_xor(p[m], off, 32);
    const int wave = tid >> 5, lane = tid & 31;
    if (lane == 0)
#pragma unroll
        for (int m = 0; m < 16; ++m) red[m][wave] = p[m];

    // router logits (exact fp32): 128 threads, one (token, expert) pair each
    if (tid < 128) {
        const int m = tid >> 3, e = tid & 7;
        float acc = br[e];
        for (int h = 0; h < HD; ++h)
            acc = fmaf(xs[m][h], Wr[(size_t)h * 8 + e], acc);
        lg[m][e] = acc;                       // TEMPERATURE == 1
    }
    __syncthreads();

    if (tid < 16) {
        const int m = tid, t = t0 + m;
        float s = 0.f;
#pragma unroll
        for (int w = 0; w < 8; ++w) s += red[m][w];
        const float conf = 1.f / (1.f + __expf(-(s + bc2[0])));
        outConf[t] = conf;
        int dk = (int)rintf(1.f + 3.f * (1.f - conf));   // round-half-even like jnp.round
        dk = dk < 1 ? 1 : (dk > 4 ? 4 : dk);

        float l[8]; float mx = -3.4e38f;
#pragma unroll
        for (int e = 0; e < 8; ++e) { l[e] = lg[m][e]; outLG[(size_t)t * 8 + e] = l[e]; mx = fmaxf(mx, l[e]); }
        float se = 0.f;
#pragma unroll
        for (int e = 0; e < 8; ++e) { l[e] = __expf(l[e] - mx); se += l[e]; }
        const float inv = 1.f / se;
#pragma unroll
        for (int e = 0; e < 8; ++e) l[e] *= inv;

        // stable top-4 (strict > keeps lowest index first on ties, like lax.top_k)
        int idx[4]; float val[4]; bool used[8] = {false,false,false,false,false,false,false,false};
#pragma unroll
        for (int k = 0; k < 4; ++k) {
            int best = 0; float bv = -1.f;
#pragma unroll
            for (int e = 0; e < 8; ++e)
                if (!used[e] && l[e] > bv) { bv = l[e]; best = e; }
            used[best] = true; idx[k] = best; val[k] = bv;
        }
#pragma unroll
        for (int k = 0; k < 4; ++k) {
            const bool valid = k < dk;
            outSW[(size_t)t * 4 + k] = valid ? val[k] : 0.f;
            outSI[(size_t)t * 4 + k] = valid ? idx[k] : 0;
        }
        const float ssum = val[0] + val[1];
        const float g0 = val[0] / ssum, g1 = val[1] / ssum;
        const int s0 = 2 * t, s1 = 2 * t + 1;
        gates[s0] = g0; gates[s1] = g1;
        const int p0 = atomicAdd(&cnt[idx[0]], 1); lists[idx[0] * NTOK + p0] = s0;
        const int p1 = atomicAdd(&cnt[idx[1]], 1); lists[idx[1] * NTOK + p1] = s1;
    }
}

// ---------------- grouped bf16 WMMA GEMM (shared by FFN1 and FFN2) -------------
// Block tile: M=32 (gathered rows), N=256, K-step=32; 8 waves = 2x4 wave grid,
// each wave owns a 16x64 output piece = four v_wmma_f32_16x16x32_bf16 per K-step
// sharing one A fragment. K loop unrolled x2 so double-buffer indices are
// literals (no per-iteration cndmask/hazard NOPs for the async LDS address).
// A tile is staged HBM/L2 -> LDS with global_load_async_to_lds_b128 (ASYNCcnt),
// B tile is transposed in-register (4k x 8n) and stored as ds_store_b64.
template<int KDIM, int NDIM, bool GATHER_TOKEN, bool RELU_BF16OUT>
__global__ __launch_bounds__(256)
void gemm_kernel(const u16* __restrict__ A,      // bf16 rows, lda = KDIM
                 const u16* __restrict__ W,      // bf16 [NE][KDIM][NDIM]
                 const float* __restrict__ bias, // [NE][NDIM]
                 const int* __restrict__ cnt,
                 const int* __restrict__ lists,
                 u16*   __restrict__ outBF,      // [NSLOT][NDIM] (FFN1 h)
                 float* __restrict__ outF)       // [NSLOT][NDIM] (FFN2 y)
{
    const int e    = blockIdx.x >> 8;          // 256 tiles / expert
    const int tile = blockIdx.x & 255;
    const int c    = cnt[e];
    const int m0   = tile * 32;
    if (m0 >= c) return;                        // uniform early-exit
    const int n0   = blockIdx.y * 256;

    __shared__ __align__(16) u16 As[2][32][40];    // K padded 32->40 (80B rows)
    __shared__ __align__(16) u16 Bs[2][256][40];   // transposed W tile [n][k], swizzled
    __shared__ int rows[32];

    const int tid = threadIdx.x;
    if (tid < 32) {
        const int p = m0 + tid;
        rows[tid] = (p < c) ? lists[e * NTOK + p] : -1;
    }
    __syncthreads();

    const int wave = tid >> 5, lane = tid & 31;
    const int wm = wave >> 2, wn = wave & 3;
    const int lrow = lane & 15, lhi = lane >> 4;
    const u16* We = W + (size_t)e * KDIM * NDIM;

    // ---- hoisted A staging state: 128 threads move the 32x32 A tile,
    //      16 B per thread per K-step, straight into LDS (no VGPR round trip).
    //      Invalid rows read row 0 (harmless; epilogue skips them).
    unsigned ldsA0 = 0u, ldsA1 = 0u;
    unsigned long long gA = 0ull;
    if (tid < 128) {
        const int r = tid >> 2, kseg = (tid & 3) * 8;
        const int slot = rows[r];
        const size_t arow = (slot >= 0) ? (size_t)(GATHER_TOKEN ? (slot >> 1) : slot) : 0;
        gA    = (unsigned long long)(size_t)(A + arow * KDIM + kseg);
        ldsA0 = (unsigned)(size_t)&As[0][r][kseg];
        ldsA1 = (unsigned)(size_t)&As[1][r][kseg];
    }
    // ---- hoisted B staging state: each thread owns 4 k x 8 n
    const int kB   = wave * 4;         // 0..28
    const u16* bSrc = We + (size_t)kB * NDIM + n0 + lane * 8;

    v8f acc0 = {}, acc1 = {}, acc2 = {}, acc3 = {};

    auto stage = [&](int k0, int buf) {
        if (tid < 128) {
            asm volatile("global_load_async_to_lds_b128 %0, %1, off"
                         :: "v"(buf ? ldsA1 : ldsA0),
                            "v"(gA + (unsigned long long)(k0 * 2))
                         : "memory");
        }
        const u16* src = bSrc + (size_t)k0 * NDIM;
        if (k0 + 68 <= KDIM)  // hint the tile after next into GL2
            __builtin_prefetch(src + (size_t)64 * NDIM, 0, 1);
        union { uint4 d; u16 s[8]; } r0, r1, r2, r3;
        r0.d = *(const uint4*)(src);
        r1.d = *(const uint4*)(src + NDIM);
        r2.d = *(const uint4*)(src + 2 * (size_t)NDIM);
        r3.d = *(const uint4*)(src + 3 * (size_t)NDIM);
        const int cB = kB >> 3, wB = kB & 7;
#pragma unroll
        for (int j = 0; j < 8; ++j) {
            const int n = lane * 8 + j;
            union { unsigned long long q; u16 s[4]; } v;
            v.s[0] = r0.s[j]; v.s[1] = r1.s[j]; v.s[2] = r2.s[j]; v.s[3] = r3.s[j];
            *(unsigned long long*)&Bs[buf][n][bofs(n, cB) + wB] = v.q;
        }
    };

    auto compute = [&](int buf) {
        Frag fa, fb[4];
        const int arow = 16 * wm + lrow;
        const int kb = 8 * lhi;
        fa.q[0] = *(const uint4*)&As[buf][arow][kb];
        fa.q[1] = *(const uint4*)&As[buf][arow][kb + 16];
#pragma unroll
        for (int s = 0; s < 4; ++s) {
            const int bn = 64 * wn + 16 * s + lrow;
            fb[s].q[0] = *(const uint4*)&Bs[buf][bn][bofs(bn, lhi)];
            fb[s].q[1] = *(const uint4*)&Bs[buf][bn][bofs(bn, lhi + 2)];
        }
        acc0 = __builtin_amdgcn_wmma_f32_16x16x32_bf16(false, fa.v, false, fb[0].v,
                                                       (short)0, acc0, false, false);
        acc1 = __builtin_amdgcn_wmma_f32_16x16x32_bf16(false, fa.v, false, fb[1].v,
                                                       (short)0, acc1, false, false);
        acc2 = __builtin_amdgcn_wmma_f32_16x16x32_bf16(false, fa.v, false, fb[2].v,
                                                       (short)0, acc2, false, false);
        acc3 = __builtin_amdgcn_wmma_f32_16x16x32_bf16(false, fa.v, false, fb[3].v,
                                                       (short)0, acc3, false, false);
    };

    stage(0, 0);
    asm volatile("s_wait_asynccnt 0x0" ::: "memory");
    __syncthreads();

    for (int k0 = 0; k0 < KDIM; k0 += 64) {      // KDIM is a multiple of 64
        stage(k0 + 32, 1);                        // always in range (k0 <= KDIM-64)
        compute(0);
        asm volatile("s_wait_asynccnt 0x0" ::: "memory");
        __syncthreads();
        if (k0 + 64 < KDIM) stage(k0 + 64, 0);
        compute(1);
        asm volatile("s_wait_asynccnt 0x0" ::: "memory");
        __syncthreads();
    }

    // epilogue: C/D layout — VGPR r, lanes 0-15 -> M=r, lanes 16-31 -> M=r+8
    const float* be = bias + (size_t)e * NDIM;
#pragma unroll
    for (int r = 0; r < 8; ++r) {
        const int m = 16 * wm + r + 8 * lhi;
        const int slot = rows[m];
        if (slot < 0) continue;
        const int colb = n0 + 64 * wn + lrow;
        float v0 = acc0[r] + be[colb];
        float v1 = acc1[r] + be[colb + 16];
        float v2 = acc2[r] + be[colb + 32];
        float v3 = acc3[r] + be[colb + 48];
        if (RELU_BF16OUT) {
            v0 = fmaxf(v0, 0.f); v1 = fmaxf(v1, 0.f);
            v2 = fmaxf(v2, 0.f); v3 = fmaxf(v3, 0.f);
            u16* o = outBF + (size_t)slot * NDIM + colb;
            o[0]  = f2bf(v0);
            o[16] = f2bf(v1);
            o[32] = f2bf(v2);
            o[48] = f2bf(v3);
        } else {
            float* o = outF + (size_t)slot * NDIM + colb;
            o[0]  = v0;
            o[16] = v1;
            o[32] = v2;
            o[48] = v3;
        }
    }
}

// ---------------- combine: out[t] = g0*y[2t] + g1*y[2t+1] (deterministic) ------
__global__ __launch_bounds__(256)
void combine_kernel(const float* __restrict__ y, const float* __restrict__ gates,
                    float* __restrict__ out)
{
    const int i = blockIdx.x * 256 + threadIdx.x;   // float4 group id, N*H/4 total
    const int t = i >> 8;                            // H/4 = 256 groups per token
    const int g4 = i & 255;
    const float g0 = gates[2 * t], g1 = gates[2 * t + 1];
    const float4 y0 = ((const float4*)(y + (size_t)(2 * t) * HD))[g4];
    const float4 y1 = ((const float4*)(y + (size_t)(2 * t + 1) * HD))[g4];
    float4 o;
    o.x = g0 * y0.x + g1 * y1.x;
    o.y = g0 * y0.y + g1 * y1.y;
    o.z = g0 * y0.z + g1 * y1.z;
    o.w = g0 * y0.w + g1 * y1.w;
    ((float4*)out)[i] = o;
}

// ---------------- launch ----------------
extern "C" void kernel_launch(void* const* d_in, const int* in_sizes, int n_in,
                              void* d_out, int out_size, void* d_ws, size_t ws_size,
                              hipStream_t stream) {
    (void)in_sizes; (void)n_in; (void)out_size; (void)ws_size;

    const float* x   = (const float*)d_in[0];
    const float* Wr  = (const float*)d_in[1];
    const float* br  = (const float*)d_in[2];
    const float* Wc1 = (const float*)d_in[3];
    const float* bc1 = (const float*)d_in[4];
    const float* Wc2 = (const float*)d_in[5];
    const float* bc2 = (const float*)d_in[6];
    const float* W1  = (const float*)d_in[7];
    const float* b1  = (const float*)d_in[8];
    const float* W2  = (const float*)d_in[9];
    const float* b2  = (const float*)d_in[10];

    char* ws = (char*)d_ws;
    u16*   xbf   = (u16*)  (ws + OFF_XBF);
    u16*   w1bf  = (u16*)  (ws + OFF_W1BF);
    u16*   w2bf  = (u16*)  (ws + OFF_W2BF);
    u16*   hbuf  = (u16*)  (ws + OFF_HBUF);
    float* ybuf  = (float*)(ws + OFF_YBUF);
    float* gates = (float*)(ws + OFF_GATE);
    int*   cnt   = (int*)  (ws + OFF_CNT);
    int*   lists = (int*)  (ws + OFF_LIST);

    float* out    = (float*)d_out;
    float* outSW  = out + (size_t)NTOK * HD;
    int*   outSI  = (int*)(outSW + (size_t)NTOK * 4);
    float* outConf= (float*)(outSI + (size_t)NTOK * 4);
    float* outLG  = outConf + NTOK;

    init_cnt_kernel<<<1, 32, 0, stream>>>(cnt);

    convert_bf16_kernel<<<2048, 256, 0, stream>>>(x,  xbf,  (size_t)NTOK * HD / 4);
    convert_bf16_kernel<<<4096, 256, 0, stream>>>(W1, w1bf, (size_t)NE * HD * ID / 4);
    convert_bf16_kernel<<<4096, 256, 0, stream>>>(W2, w2bf, (size_t)NE * ID * HD / 4);

    routing_kernel<<<NTOK / 16, 256, 0, stream>>>(x, Wr, br, Wc1, bc1, Wc2, bc2,
                                                  outSW, outSI, outConf, outLG,
                                                  gates, cnt, lists);

    // FFN1: h = relu(x @ W1[e] + b1[e])  -> bf16 hbuf   (K=1024, N=4096)
    gemm_kernel<HD, ID, true, true>
        <<<dim3(NE * TILES_PER_E, ID / 256), 256, 0, stream>>>(
            xbf, w1bf, b1, cnt, lists, hbuf, nullptr);

    // FFN2: y = h @ W2[e] + b2[e]        -> f32 ybuf    (K=4096, N=1024)
    gemm_kernel<ID, HD, false, false>
        <<<dim3(NE * TILES_PER_E, HD / 256), 256, 0, stream>>>(
            hbuf, w2bf, b2, cnt, lists, nullptr, ybuf);

    combine_kernel<<<(NTOK * HD / 4) / 256, 256, 0, stream>>>(ybuf, gates, out);
}
